// L1_att_block_11656541241667
// MI455X (gfx1250) — matrix operand order repaired and checked by reading, not compiled, not run on hardware
//
#include <hip/hip_runtime.h>
#include <hip/hip_bf16.h>

typedef __attribute__((ext_vector_type(16))) _Float16 v16h;
typedef __attribute__((ext_vector_type(8)))  _Float16 h8;
typedef __attribute__((ext_vector_type(4)))  _Float16 h4;
typedef __attribute__((ext_vector_type(8)))  float    v8f;
typedef __attribute__((ext_vector_type(4)))  float    f4;

#define SPW   4                 // samples per workgroup
#define ROWS  (SPW * 28)        // 112 rows of X per workgroup (7 exact 16-row tiles)
#define KPAD  32                // K=28 padded to 32 -> one 16x16x32 WMMA per tile
#define NPAD  32                // output/N dimension padded to 32 (cols 28..31 = 0)
#define MTILES 7
#define NTILES 2
#define TILES_PER_MAT (MTILES * NTILES)   // 14
#define TOTAL_TILES   (3 * TILES_PER_MAT) // 42 (q,k,v)

__global__ __launch_bounds__(128)
void l1attn_fused_kernel(const float* __restrict__ x,
                         const float* __restrict__ Wq, const float* __restrict__ bq,
                         const float* __restrict__ Wk, const float* __restrict__ bk,
                         const float* __restrict__ Wv, const float* __restrict__ bv,
                         const float* __restrict__ W1, const float* __restrict__ b1,
                         const float* __restrict__ W2, const float* __restrict__ b2,
                         float* __restrict__ out, int bs)
{
    __shared__ alignas(16) _Float16 xA[ROWS * KPAD];       // x rows, f16, K padded
    __shared__ alignas(16) _Float16 Wf[3 * NPAD * KPAD];   // W rows, f16, N+K padded
    __shared__ float    biasS[3 * NPAD];                   // padded biases
    __shared__ alignas(16) float qkv[3 * ROWS * NPAD];     // q,k,v results, stride 32
    __shared__ float    l1s[ROWS];
    __shared__ float    hs[SPW * 20];

    const int tid = threadIdx.x;
    const int s0  = blockIdx.x * SPW;

    // ---------- phase 0a: stage x (float4 -> 4xf16, ds_store_b64) -----------
    for (int it = tid; it < ROWS * 8; it += 128) {
        int row = it >> 3, g = it & 7;
        h4 o = (h4)(_Float16)0.f;
        if (g < 7) {
            int sG = s0 + row / 28;
            if (sG < bs) {
                f4 v = *(const f4*)&x[sG * 784 + (row % 28) * 28 + g * 4];
                o[0] = (_Float16)v[0]; o[1] = (_Float16)v[1];
                o[2] = (_Float16)v[2]; o[3] = (_Float16)v[3];
            }
        }
        *(h4*)&xA[row * KPAD + g * 4] = o;
    }

    // ---------- phase 0b: stage Wq/Wk/Wv (uniform passes) -------------------
    #pragma unroll
    for (int mat = 0; mat < 3; ++mat) {
        const float* W = (mat == 0) ? Wq : (mat == 1) ? Wk : Wv;
        for (int it = tid; it < NPAD * 8; it += 128) {   // 256 items -> 2 passes
            int n = it >> 3, g = it & 7;
            h4 o = (h4)(_Float16)0.f;
            if (g < 7 && n < 28) {
                f4 v = *(const f4*)&W[n * 28 + g * 4];
                o[0] = (_Float16)v[0]; o[1] = (_Float16)v[1];
                o[2] = (_Float16)v[2]; o[3] = (_Float16)v[3];
            }
            *(h4*)&Wf[(mat * NPAD + n) * KPAD + g * 4] = o;
        }
    }
    if (tid < 3 * NPAD) {                 // 96 < 128: single pass
        int mat = tid >> 5, n = tid & 31;
        const float* B = (mat == 0) ? bq : (mat == 1) ? bk : bv;
        biasS[tid] = (n < 28) ? B[n] : 0.f;
    }
    __syncthreads();

    // ---------- phase 1: WMMA projections (q,k,v) ---------------------------
    const int wave = tid >> 5;
    const int lane = tid & 31;
    const int l    = lane & 15;
    const int hi   = lane >> 4;

    for (int T = wave; T < TOTAL_TILES; T += 4) {   // wave-uniform loop
        int mat   = T / TILES_PER_MAT;
        int rem   = T - mat * TILES_PER_MAT;
        int mtile = rem >> 1;
        int ntile = rem & 1;
        int rowBase = mtile * 16;
        int N       = ntile * 16 + l;               // 0..31, always in padded range

        // A: lane row = rowBase+l; e=0..7 <-> K=hi*8+e, e=8..15 <-> K=16+hi*8+(e-8)
        const _Float16* arow = &xA[(rowBase + l) * KPAD];
        h8 a0 = *(const h8*)(arow + (hi << 3));
        h8 a1 = *(const h8*)(arow + 16 + (hi << 3));
        v16h a = __builtin_shufflevector(a0, a1, 0,1,2,3,4,5,6,7,8,9,10,11,12,13,14,15);

        // B: lane column N; e=0..7 <-> K=hi*8+e, e=8..15 <-> K=16+hi*8+(e-8)
        const _Float16* brow = &Wf[(mat * NPAD + N) * KPAD + (hi << 3)];
        h8 b0 = *(const h8*)(brow);
        h8 b1 = *(const h8*)(brow + 16);
        v16h b = __builtin_shufflevector(b0, b1, 0,1,2,3,4,5,6,7,8,9,10,11,12,13,14,15);

        // C: bias broadcast down each column (C[M][N] = bias[N])
        float bc = biasS[mat * NPAD + N];
        v8f c;
        #pragma unroll
        for (int r = 0; r < 8; ++r) c[r] = bc;

        v8f d = __builtin_amdgcn_wmma_f32_16x16x32_f16(
                    false, a, false, b, (short)0, c, false, false);

        // scatter D: lane(l,hi) VGPR r -> row rowBase + r + 8*hi, col N (padded)
        float* drow = &qkv[mat * ROWS * NPAD + rowBase * NPAD + N];
        #pragma unroll
        for (int r = 0; r < 8; ++r)
            drow[(r + (hi << 3)) * NPAD] = d[r];
    }
    __syncthreads();

    // ---------- phase 2: L1 attention + row-dot -----------------------------
    // l1[m] = sum_n v[m,n] * coeff(n),  coeff = (n==m) ? 1 : scale*sum_w|q[n,w]-k[m,w]|
    if (tid < ROWS) {
        int sample = tid / 28;
        int m      = tid - sample * 28;
        int base   = sample * 28;
        const float* qb = &qkv[0 * ROWS * NPAD + base * NPAD];
        const float* kb = &qkv[1 * ROWS * NPAD + (base + m) * NPAD];
        const float* vb = &qkv[2 * ROWS * NPAD + (base + m) * NPAD];

        // k row stays in 7 native f4 vectors: only compile-time indexed -> no repack
        f4 kq[7];
        #pragma unroll
        for (int g = 0; g < 7; ++g) kq[g] = *(const f4*)&kb[g * 4];

        const float scale = -0.1889822365046136f;   // -1/sqrt(28)
        float l1 = 0.f;
        for (int n = 0; n < 28; ++n) {
            const float* qn = &qb[n * NPAD];
            float s = 0.f;
            #pragma unroll
            for (int g = 0; g < 7; ++g) {
                f4 t = *(const f4*)&qn[g * 4];
                s += fabsf(t[0] - kq[g][0]) + fabsf(t[1] - kq[g][1])
                   + fabsf(t[2] - kq[g][2]) + fabsf(t[3] - kq[g][3]);
            }
            float coeff = (n == m) ? 1.0f : scale * s;   // v_cndmask, no branch
            l1 += vb[n] * coeff;                          // scalar ds_load_b32
        }
        l1s[tid] = l1;
    }
    __syncthreads();

    // ---------- phase 3: MLP hidden (28 -> 20, relu) ------------------------
    if (tid < SPW * 20) {
        int sample = tid / 20, j = tid - sample * 20;
        const float* w  = &W1[j * 28];
        const float* li = &l1s[sample * 28];
        float acc = b1[j];
        #pragma unroll 7
        for (int i = 0; i < 28; ++i) acc += li[i] * w[i];
        hs[tid] = acc > 0.f ? acc : 0.f;
    }
    __syncthreads();

    // ---------- phase 4: MLP out (20 -> 10) ---------------------------------
    if (tid < SPW * 10) {
        int sample = tid / 10, j = tid - sample * 10;
        int sG = s0 + sample;
        if (sG < bs) {
            const float* w = &W2[j * 20];
            const float* h = &hs[sample * 20];
            float acc = b2[j];
            #pragma unroll 5
            for (int i = 0; i < 20; ++i) acc += h[i] * w[i];
            out[sG * 10 + j] = acc;
        }
    }
}

extern "C" void kernel_launch(void* const* d_in, const int* in_sizes, int n_in,
                              void* d_out, int out_size, void* d_ws, size_t ws_size,
                              hipStream_t stream) {
    const float* x  = (const float*)d_in[0];
    const float* Wq = (const float*)d_in[1];
    const float* bq = (const float*)d_in[2];
    const float* Wk = (const float*)d_in[3];
    const float* bk = (const float*)d_in[4];
    const float* Wv = (const float*)d_in[5];
    const float* bv = (const float*)d_in[6];
    const float* W1 = (const float*)d_in[7];
    const float* b1 = (const float*)d_in[8];
    const float* W2 = (const float*)d_in[9];
    const float* b2 = (const float*)d_in[10];

    int bs   = in_sizes[0] / 784;          // x is (bs, 1, 28, 28)
    int grid = (bs + SPW - 1) / SPW;

    hipLaunchKernelGGL(l1attn_fused_kernel, dim3(grid), dim3(128), 0, stream,
                       x, Wq, bq, Wk, bk, Wv, bv, W1, b1, W2, b2,
                       (float*)d_out, bs);
}
